// SubtokenMerger_52183852647007
// MI455X (gfx1250) — compile-verified, matrix operand order checked by reading, not run on hardware
//
#include <hip/hip_runtime.h>

// Problem constants (from the reference)
#define BB 32
#define SS 1024
#define HH 1024
#define WW 128
#define KK 6
#define NH 16
#define HD 64
#define GWIN 8                 // windows per workgroup
#define ROWS (GWIN * KK)       // 48 x-rows per WG -> exactly 3 M tiles of 16
#define KSTEPS (HH / 32)       // 32 K-steps of the bf16 WMMA

typedef __bf16 bf16_t;
typedef __attribute__((ext_vector_type(8)))  bf16_t v8bf;
typedef __attribute__((ext_vector_type(16))) bf16_t v16bf;
typedef __attribute__((ext_vector_type(8)))  float  v8f;
typedef __attribute__((ext_vector_type(4)))  float  v4f;

__device__ __forceinline__ v8f zero8() {
  v8f z;
#pragma unroll
  for (int i = 0; i < 8; ++i) z[i] = 0.f;
  return z;
}

// Load one 16x32 bf16 WMMA operand fragment for this lane.
// Per ISA 7.12.2: lane row = lane%16; K halves [kh..kh+8) U [kh+16..kh+24),
// kh = (lane<16 ? 0 : 8).  p must point at (row base + k0 + kh).
__device__ __forceinline__ v16bf load_frag(const bf16_t* p) {
  v8bf lo = *reinterpret_cast<const v8bf*>(p);       // 16B chunk
  v8bf hi = *reinterpret_cast<const v8bf*>(p + 16);  // 16B chunk
  v16bf r;
#pragma unroll
  for (int i = 0; i < 8; ++i) { r[i] = lo[i]; r[i + 8] = hi[i]; }
  return r;
}

// ---------------- K1: fp32 weight -> bf16 (first 2H rows) ----------------
__global__ void wconv_kernel(const float* __restrict__ w, bf16_t* __restrict__ wb) {
  const int r = blockIdx.x;           // 0..2047
  const int c = threadIdx.x * 4;      // 256 threads * 4 = 1024 cols
  v4f v = *reinterpret_cast<const v4f*>(w + (size_t)r * HH + c);
  bf16_t* d = wb + (size_t)r * HH + c;
  d[0] = (bf16_t)v[0]; d[1] = (bf16_t)v[1]; d[2] = (bf16_t)v[2]; d[3] = (bf16_t)v[3];
}

// ---------------- K2: bulk copy emb -> out (streamed) --------------------
__global__ void copy_kernel(const v4f* __restrict__ src, v4f* __restrict__ dst, long n4) {
  long i = (long)blockIdx.x * blockDim.x + threadIdx.x;
  const long stride = (long)gridDim.x * blockDim.x;
  for (; i < n4; i += stride) {
    v4f v = src[i];                        // RT load: warms L2 for the gather
    __builtin_nontemporal_store(v, dst + i); // NT store: don't pollute L2
  }
}

// ---------------- K3: fused gather + QK GEMM + attention + merge ---------
__global__ void __launch_bounds__(256)
merge_kernel(const float* __restrict__ emb,
             const bf16_t* __restrict__ wb,
             const float* __restrict__ bias,
             const int* __restrict__ sidx,
             const unsigned char* __restrict__ smask,
             float* __restrict__ out)
{
  extern __shared__ __align__(16) char smem[];
  bf16_t* Xs      = reinterpret_cast<bf16_t*>(smem);                       // [48][1024] bf16 = 96KB
  float*  stage   = reinterpret_cast<float*>(smem + ROWS * HH * 2);        // 8 waves * (q+k)[48][64] f32 = 192KB
  float*  scoresL = stage + 8 * 2 * ROWS * HD;                             // [8][16][6][6] f32 = 18KB
  float*  mvals   = scoresL + GWIN * NH * KK * KK;                         // [48]
  int*    idxsL   = reinterpret_cast<int*>(mvals + ROWS);                  // [48]
  float*  contribL= reinterpret_cast<float*>(idxsL + ROWS);                // [48]

  const int tid  = threadIdx.x;
  const int lane = tid & 31;
  const int wv   = tid >> 5;                 // 8 waves
  const int wg0  = blockIdx.x * GWIN;

  // ---- Phase 1a: indices / masks ----
  if (tid < ROWS) {
    const int lw = tid / KK, k = tid % KK;
    const int gw = wg0 + lw;
    const int b = gw >> 7, w = gw & (WW - 1);
    const size_t base = (((size_t)b * WW) + w) * KK + k;
    idxsL[tid] = sidx[base];
    mvals[tid] = smask[base] ? 1.0f : 0.0f;
  }
  __syncthreads();

  // ---- Phase 1b: gather x = mask * emb[b, idx] -> bf16 in LDS ----
  for (int e = tid; e < ROWS * 256; e += 256) {
    const int r  = e >> 8;
    const int c4 = (e & 255) << 2;
    const int gw = wg0 + (r / KK);
    const int b  = gw >> 7;
    v4f v = {0.f, 0.f, 0.f, 0.f};
    if (mvals[r] != 0.f)
      v = *reinterpret_cast<const v4f*>(emb + ((size_t)b * SS + idxsL[r]) * HH + c4);
    bf16_t* d = Xs + (size_t)r * HH + c4;
    d[0] = (bf16_t)v[0]; d[1] = (bf16_t)v[1]; d[2] = (bf16_t)v[2]; d[3] = (bf16_t)v[3];
  }
  __syncthreads();

  // ---- Phase 2: per-wave Q/K projection (WMMA) + scores, 2 heads/wave ----
  float* qst = stage + wv * (2 * ROWS * HD);
  float* kst = qst + ROWS * HD;
  const int nloc  = lane & 15;
  const int khalf = (lane < 16) ? 0 : 8;

  for (int hh = 0; hh < 2; ++hh) {
    const int h = wv * 2 + hh;
    for (int nt = 0; nt < 8; ++nt) {            // 4 q N-tiles then 4 k N-tiles
      const int wrow0 = (nt < 4) ? (h * HD + nt * 16)
                                 : (HH + h * HD + (nt - 4) * 16);
      float* st = (nt < 4) ? qst : kst;
      const int col0 = (nt & 3) * 16;
      const bf16_t* wrowp = wb + (size_t)(wrow0 + nloc) * HH + khalf;
      const bf16_t* arow0 = Xs + (size_t)(0 * 16 + nloc) * HH + khalf;
      const bf16_t* arow1 = Xs + (size_t)(1 * 16 + nloc) * HH + khalf;
      const bf16_t* arow2 = Xs + (size_t)(2 * 16 + nloc) * HH + khalf;

      v8f acc0 = zero8(), acc1 = zero8(), acc2 = zero8();
      for (int kk = 0; kk < KSTEPS; ++kk) {
        const int k0 = kk * 32;
        const v16bf bfrag = load_frag(wrowp + k0);   // B loaded once, reused 3x
        v16bf a0 = load_frag(arow0 + k0);
        acc0 = __builtin_amdgcn_wmma_f32_16x16x32_bf16(false, a0, false, bfrag,
                                                       (short)0, acc0, false, false);
        v16bf a1 = load_frag(arow1 + k0);
        acc1 = __builtin_amdgcn_wmma_f32_16x16x32_bf16(false, a1, false, bfrag,
                                                       (short)0, acc1, false, false);
        v16bf a2 = load_frag(arow2 + k0);
        acc2 = __builtin_amdgcn_wmma_f32_16x16x32_bf16(false, a2, false, bfrag,
                                                       (short)0, acc2, false, false);
      }
      // C/D layout: lane -> col nloc; VGPR r -> row r (+8 for lanes>=16)
      const float bj = bias[wrow0 + nloc];
      const int rofs = (lane < 16) ? 0 : 8;
#pragma unroll
      for (int r = 0; r < 8; ++r) {
        st[(0 * 16 + rofs + r) * HD + col0 + nloc] = acc0[r] + bj;
        st[(1 * 16 + rofs + r) * HD + col0 + nloc] = acc1[r] + bj;
        st[(2 * 16 + rofs + r) * HD + col0 + nloc] = acc2[r] + bj;
      }
    }
    asm volatile("" ::: "memory");   // same-wave LDS is in-order; just fence the compiler

    // scores[lw][h][i][j] = q_i . k_j / 8 + mask_i*mask_j
    for (int t = lane; t < GWIN * KK * KK; t += 32) {
      const int lw = t / 36, rem = t % 36, i = rem / 6, j = rem % 6;
      const float* qr = qst + (lw * KK + i) * HD;
      const float* kr = kst + (lw * KK + j) * HD;
      float d = 0.f;
#pragma unroll 8
      for (int dd = 0; dd < HD; ++dd) d += qr[dd] * kr[dd];
      const float pair = mvals[lw * KK + i] * mvals[lw * KK + j];
      scoresL[((lw * NH + h) * KK + i) * KK + j] = d * 0.125f + pair;
    }
    asm volatile("" ::: "memory");
  }
  __syncthreads();

  // ---- Phase 3: softmax over j, mean over heads, contrib (wave wv = window wv) ----
  const int lw = wv;
  if (lane == 0) {
    float wm[KK * KK];
#pragma unroll
    for (int i = 0; i < KK * KK; ++i) wm[i] = 0.f;
    for (int h = 0; h < NH; ++h) {
      const float* sp = scoresL + ((lw * NH + h) * KK) * KK;
      for (int i = 0; i < KK; ++i) {
        float s[KK], e[KK], m = -1e30f, sum = 0.f;
#pragma unroll
        for (int j = 0; j < KK; ++j) { s[j] = sp[i * KK + j]; m = fmaxf(m, s[j]); }
#pragma unroll
        for (int j = 0; j < KK; ++j) { e[j] = __expf(s[j] - m); sum += e[j]; }
        const float inv = 1.f / sum;
#pragma unroll
        for (int j = 0; j < KK; ++j) wm[i * KK + j] += e[j] * inv;
      }
    }
    float c[KK], csum = 0.f;
#pragma unroll
    for (int j = 0; j < KK; ++j) {
      float acc = 0.f;
#pragma unroll
      for (int i = 0; i < KK; ++i)
        acc += mvals[lw * KK + i] * mvals[lw * KK + j] * wm[i * KK + j] * (1.0f / NH);
      c[j] = acc; csum += acc;
    }
    const float invn = 1.f / (csum + 1e-8f);
#pragma unroll
    for (int j = 0; j < KK; ++j) contribL[lw * KK + j] = c[j] * invn;
  }
  asm volatile("" ::: "memory");     // same-wave readers below

  // ---- Phase 4: zero merged rows; write unified at first valid idx ----
  {
    const int gw = wg0 + lw;
    const int b = gw >> 7;
    float cw[KK], mv[KK]; int ridx[KK];
#pragma unroll
    for (int k = 0; k < KK; ++k) {
      cw[k] = contribL[lw * KK + k];
      mv[k] = mvals[lw * KK + k];
      ridx[k] = idxsL[lw * KK + k];
    }
#pragma unroll
    for (int k = 0; k < KK; ++k) {
      if (mv[k] == 0.f) continue;
      float* dst = out + ((size_t)b * SS + ridx[k]) * HH;
      if (k == 0) {
        for (int c4 = lane; c4 < HH / 4; c4 += 32) {
          v4f acc = {0.f, 0.f, 0.f, 0.f};
#pragma unroll
          for (int kk = 0; kk < KK; ++kk) {
            if (mv[kk] != 0.f) {
              v4f v = *(reinterpret_cast<const v4f*>(emb) +
                        ((size_t)b * SS + ridx[kk]) * (HH / 4) + c4);
              acc += cw[kk] * v;
            }
          }
          *(reinterpret_cast<v4f*>(dst) + c4) = acc;
        }
      } else {
        const v4f z = {0.f, 0.f, 0.f, 0.f};
        for (int c4 = lane; c4 < HH / 4; c4 += 32)
          *(reinterpret_cast<v4f*>(dst) + c4) = z;
      }
    }
  }
}

extern "C" void kernel_launch(void* const* d_in, const int* in_sizes, int n_in,
                              void* d_out, int out_size, void* d_ws, size_t ws_size,
                              hipStream_t stream) {
  (void)in_sizes; (void)n_in; (void)out_size; (void)ws_size;
  const float*         emb   = (const float*)d_in[0];
  const float*         wfp   = (const float*)d_in[1];
  const float*         bias  = (const float*)d_in[2];
  const int*           sidx  = (const int*)d_in[3];
  const unsigned char* smask = (const unsigned char*)d_in[4];  // numpy bool = 1 byte
  float*   out = (float*)d_out;
  bf16_t*  wb  = (bf16_t*)d_ws;                                // 4 MB scratch

  // K1: fp32 -> bf16 weight (2H x H)
  wconv_kernel<<<2 * HH, 256, 0, stream>>>(wfp, wb);

  // K2: out = emb (full copy; later kernel overwrites merged rows)
  const long n4 = (long)BB * SS * HH / 4;
  copy_kernel<<<4096, 256, 0, stream>>>((const v4f*)emb, (v4f*)out, n4);

  // K3: fused kernel, 512 WGs * 256 threads, ~307 KB dynamic LDS
  const int SMEM = ROWS * HH * 2                    // Xs bf16
                 + 8 * 2 * ROWS * HD * 4            // q/k staging f32
                 + GWIN * NH * KK * KK * 4          // scores
                 + ROWS * 4 * 3;                    // mvals/idx/contrib
  hipFuncSetAttribute((const void*)merge_kernel,
                      hipFuncAttributeMaxDynamicSharedMemorySize, SMEM);
  merge_kernel<<<(BB * WW) / GWIN, 256, SMEM, stream>>>(emb, wb, bias, sidx, smask, out);
}
